// MultiHeadAttention_781684048588
// MI455X (gfx1250) — matrix operand order repaired and checked by reading, not compiled, not run on hardware
//
#include <hip/hip_runtime.h>
#include <hip/hip_bf16.h>

// ---------------------------------------------------------------------------
// MHA + FiLM block for gfx1250 (MI455X). bf16 WMMA (16x16x32) everywhere,
// f32 accumulate. GEMM operands pre-converted to bf16 in WMMA-friendly
// layouts; waves own 32x64 output strips (8 accumulators) so each loaded
// fragment feeds 2 WMMAs: inner loops are 12 b128 loads per 8 v_wmma.
// ---------------------------------------------------------------------------

typedef __attribute__((ext_vector_type(16))) __bf16        v16bf;
typedef __attribute__((ext_vector_type(8)))  float         v8f;
typedef __attribute__((ext_vector_type(4)))  unsigned int  u32x4;

union V16U {
    v16bf          v;
    unsigned short u[16];
    u32x4          q[2];
};
union V8U {
    unsigned short u[8];
    u32x4          q;
};

__device__ __forceinline__ unsigned short f2bf(float f) {
    unsigned u = __builtin_bit_cast(unsigned, f);
    u += 0x7FFFu + ((u >> 16) & 1u);          // round-to-nearest-even
    return (unsigned short)(u >> 16);
}

__device__ __forceinline__ v8f wmma_bf16(v16bf a, v16bf b, v8f c) {
    return __builtin_amdgcn_wmma_f32_16x16x32_bf16(
        false, a, false, b, (short)0, c, false, false);
}

__device__ __forceinline__ float wave_sum(float v) {
    #pragma unroll
    for (int o = 16; o > 0; o >>= 1) v += __shfl_xor(v, o, 32);
    return v;
}
__device__ __forceinline__ float wave_max(float v) {
    #pragma unroll
    for (int o = 16; o > 0; o >>= 1) v = fmaxf(v, __shfl_xor(v, o, 32));
    return v;
}

// Problem constants
#define BB   8
#define LL   1024
#define DD   512
#define HH   8
#define DKD  64

// ---------------------------------------------------------------------------
// Elementwise f32 -> bf16 (8 elems/thread, vector loads/stores).
// ---------------------------------------------------------------------------
__global__ void cvt_x_bf16(const float* __restrict__ x, unsigned short* __restrict__ y) {
    const size_t i = ((size_t)blockIdx.x * 256 + threadIdx.x) * 8;
    V8U t;
    #pragma unroll
    for (int j = 0; j < 8; ++j) t.u[j] = f2bf(x[i + j]);
    *(u32x4*)(y + i) = t.q;
}

// Weight convert + transpose: Wt[n,k] (bf16) = W[k,n] (f32).  [512x512]
__global__ void cvt_w_t(const float* __restrict__ W, unsigned short* __restrict__ Wt) {
    const int id = blockIdx.x * 256 + threadIdx.x;      // 262144 total
    const int n = id >> 9, k = id & 511;
    Wt[id] = f2bf(W[(size_t)k * 512 + n]);
}

// ---------------------------------------------------------------------------
// GEMM: Y[8192,512] = Xbf[8192,512] * Wt^T + bias   (Wt is [N=512, K=512] bf16)
// Wave -> 32 rows x 64 cols (2 M-tiles x 4 N-tiles; 8 WMMA per k-step).
// Block 256 (8 waves) -> 256 rows x 64 cols. grid (512/64=8, 8192/256=32).
// MODE 0: Y f32 row-major. MODE 1: Y bf16 [B,H,L,64]. MODE 2: Y bf16 [B,H,64,L].
// ---------------------------------------------------------------------------
template <int MODE>
__global__ void gemm512_wmma(const unsigned short* __restrict__ Xbf,
                             const unsigned short* __restrict__ Wt,
                             const float* __restrict__ bias,
                             void* __restrict__ Yv) {
    const int lane = threadIdx.x & 31;
    const int wave = threadIdx.x >> 5;
    const int m0 = blockIdx.y * 256 + wave * 32;
    const int n0 = blockIdx.x * 64;
    const int row = lane & 15;
    const int hi  = lane >> 4;

    const unsigned short* Ap0 = Xbf + (size_t)(m0 + row) * 512 + hi * 8;
    const unsigned short* Ap1 = Xbf + (size_t)(m0 + 16 + row) * 512 + hi * 8;
    const unsigned short* Bp[4];
    #pragma unroll
    for (int s = 0; s < 4; ++s)
        Bp[s] = Wt + (size_t)(n0 + s * 16 + row) * 512 + hi * 8;

    v8f acc0[4] = {{}, {}, {}, {}};
    v8f acc1[4] = {{}, {}, {}, {}};
    for (int kk = 0; kk < 512; kk += 32) {
        __builtin_prefetch(Ap0 + kk + 64, 0, 1);
        __builtin_prefetch(Ap1 + kk + 64, 0, 1);
        V16U a0, a1;
        a0.q[0] = *(const u32x4*)(Ap0 + kk);
        a0.q[1] = *(const u32x4*)(Ap0 + kk + 16);
        a1.q[0] = *(const u32x4*)(Ap1 + kk);
        a1.q[1] = *(const u32x4*)(Ap1 + kk + 16);
        #pragma unroll
        for (int s = 0; s < 4; ++s) {
            V16U b;
            b.q[0] = *(const u32x4*)(Bp[s] + kk);
            b.q[1] = *(const u32x4*)(Bp[s] + kk + 16);
            acc0[s] = wmma_bf16(a0.v, b.v, acc0[s]);
            acc1[s] = wmma_bf16(a1.v, b.v, acc1[s]);
        }
    }

    #pragma unroll
    for (int s = 0; s < 4; ++s) {
        const int n = n0 + s * 16 + row;
        const float bsv = bias[n];
        #pragma unroll
        for (int t = 0; t < 2; ++t) {
            #pragma unroll
            for (int v = 0; v < 8; ++v) {
                const int m = m0 + t * 16 + v + (hi << 3);
                const float val = (t ? acc1[s][v] : acc0[s][v]) + bsv;
                if (MODE == 0) {
                    ((float*)Yv)[(size_t)m * 512 + n] = val;
                } else {
                    const unsigned short ov = f2bf(val);
                    const int bq = m >> 10, l = m & 1023;
                    const int h = n >> 6,  d = n & 63;
                    if (MODE == 1)
                        ((unsigned short*)Yv)[((((size_t)bq * HH + h) * LL + l) << 6) + d] = ov;
                    else
                        ((unsigned short*)Yv)[((((size_t)bq * HH + h) * DKD + d) << 10) + l] = ov;
                }
            }
        }
    }
}

// ---------------------------------------------------------------------------
// scores = Q K^T / 8 -> attn region (pre-softmax).
// Wave -> 32 q-rows x 64 keys (2x4 tiles; both fragments reused).
// grid (B*H=64, L/32=32, 2), block 256.
// ---------------------------------------------------------------------------
__global__ void attn_scores_wmma(const unsigned short* __restrict__ Qbf,
                                 const unsigned short* __restrict__ Kbf,
                                 float* __restrict__ attn) {
    const int lane = threadIdx.x & 31;
    const int wave = threadIdx.x >> 5;
    const int bh = blockIdx.x;
    const int q0 = blockIdx.y * 32;
    const int c0 = (blockIdx.z * 8 + wave) * 64;
    const int row = lane & 15;
    const int hi  = lane >> 4;

    const unsigned short* Qr0 = Qbf + ((size_t)bh * LL + q0 + row) * DKD + hi * 8;
    const unsigned short* Qr1 = Qbf + ((size_t)bh * LL + q0 + 16 + row) * DKD + hi * 8;
    const unsigned short* Kr[4];
    #pragma unroll
    for (int s = 0; s < 4; ++s)
        Kr[s] = Kbf + ((size_t)bh * LL + c0 + s * 16 + row) * DKD + hi * 8;

    v8f acc0[4] = {{}, {}, {}, {}};
    v8f acc1[4] = {{}, {}, {}, {}};
    #pragma unroll
    for (int kk = 0; kk < 64; kk += 32) {
        V16U a0, a1;
        a0.q[0] = *(const u32x4*)(Qr0 + kk);
        a0.q[1] = *(const u32x4*)(Qr0 + kk + 16);
        a1.q[0] = *(const u32x4*)(Qr1 + kk);
        a1.q[1] = *(const u32x4*)(Qr1 + kk + 16);
        #pragma unroll
        for (int s = 0; s < 4; ++s) {
            V16U b;
            b.q[0] = *(const u32x4*)(Kr[s] + kk);
            b.q[1] = *(const u32x4*)(Kr[s] + kk + 16);
            acc0[s] = wmma_bf16(a0.v, b.v, acc0[s]);
            acc1[s] = wmma_bf16(a1.v, b.v, acc1[s]);
        }
    }

    float* out = attn + ((size_t)bh * LL + q0) * LL + c0;
    #pragma unroll
    for (int s = 0; s < 4; ++s)
        #pragma unroll
        for (int t = 0; t < 2; ++t)
            #pragma unroll
            for (int v = 0; v < 8; ++v) {
                const int m = t * 16 + v + (hi << 3);
                out[(size_t)m * LL + s * 16 + row] =
                    (t ? acc1[s][v] : acc0[s][v]) * 0.125f;
            }
}

// ---------------------------------------------------------------------------
// Row softmax in place over attn (65536 rows x 1024). One wave per row.
// ---------------------------------------------------------------------------
__global__ void softmax_rows_k(float* __restrict__ attn) {
    const int lane = threadIdx.x & 31;
    const int wave = threadIdx.x >> 5;
    const size_t r = (size_t)blockIdx.x * 8 + wave;
    float* p = attn + r * LL;

    float vals[32];
    float mx = -3.4e38f;
    #pragma unroll
    for (int i = 0; i < 32; ++i) { vals[i] = p[i * 32 + lane]; mx = fmaxf(mx, vals[i]); }
    mx = wave_max(mx);
    float s = 0.f;
    #pragma unroll
    for (int i = 0; i < 32; ++i) { vals[i] = __expf(vals[i] - mx); s += vals[i]; }
    s = wave_sum(s);
    const float inv = 1.f / s;
    #pragma unroll
    for (int i = 0; i < 32; ++i) p[i * 32 + lane] = vals[i] * inv;
}

// ---------------------------------------------------------------------------
// o = attn @ V  per head ([1024x1024] x [1024x64]); Vt is [B,H,64,L] bf16.
// Wave -> 32 q-rows x all 64 d (2x4 tiles; conversion + V fragments reused).
// grid (B*H=64, L/256=4), block 256. o written f32 [B*L, 512].
// ---------------------------------------------------------------------------
__global__ void attn_v_wmma(const float* __restrict__ attn,
                            const unsigned short* __restrict__ Vt,
                            float* __restrict__ o) {
    const int lane = threadIdx.x & 31;
    const int wave = threadIdx.x >> 5;
    const int bh = blockIdx.x;
    const int q0 = blockIdx.y * 256 + wave * 32;
    const int row = lane & 15;
    const int hi  = lane >> 4;

    const float* Ar0 = attn + ((size_t)bh * LL + q0 + row) * LL + hi * 8;
    const float* Ar1 = attn + ((size_t)bh * LL + q0 + 16 + row) * LL + hi * 8;
    const unsigned short* Br[4];
    #pragma unroll
    for (int s = 0; s < 4; ++s)
        Br[s] = Vt + ((size_t)bh * DKD + s * 16 + row) * LL + hi * 8;

    v8f acc0[4] = {{}, {}, {}, {}};
    v8f acc1[4] = {{}, {}, {}, {}};
    for (int kk = 0; kk < 1024; kk += 32) {
        __builtin_prefetch(Ar0 + kk + 64, 0, 1);
        __builtin_prefetch(Ar1 + kk + 64, 0, 1);
        V16U a0, a1;
        #pragma unroll
        for (int i = 0; i < 8; ++i) a0.u[i] = f2bf(Ar0[kk + i]);
        #pragma unroll
        for (int i = 0; i < 8; ++i) a0.u[8 + i] = f2bf(Ar0[kk + 16 + i]);
        #pragma unroll
        for (int i = 0; i < 8; ++i) a1.u[i] = f2bf(Ar1[kk + i]);
        #pragma unroll
        for (int i = 0; i < 8; ++i) a1.u[8 + i] = f2bf(Ar1[kk + 16 + i]);
        #pragma unroll
        for (int s = 0; s < 4; ++s) {
            V16U b;
            b.q[0] = *(const u32x4*)(Br[s] + kk);
            b.q[1] = *(const u32x4*)(Br[s] + kk + 16);
            acc0[s] = wmma_bf16(a0.v, b.v, acc0[s]);
            acc1[s] = wmma_bf16(a1.v, b.v, acc1[s]);
        }
    }

    const int bq = bh >> 3, h = bh & 7;
    #pragma unroll
    for (int s = 0; s < 4; ++s) {
        const int n = s * 16 + row;
        #pragma unroll
        for (int t = 0; t < 2; ++t)
            #pragma unroll
            for (int v = 0; v < 8; ++v) {
                const int m = q0 + t * 16 + v + (hi << 3);
                o[((size_t)bq * LL + m) * DD + h * DKD + n] =
                    (t ? acc1[s][v] : acc0[s][v]);
            }
    }
}

// ---------------------------------------------------------------------------
// FiLM MLP: film[b, 0:1024] = silu(latent[b]) @ Ws1 + bs1   (tiny)
// ---------------------------------------------------------------------------
__global__ void film_mlp_k(const float* __restrict__ latent,
                           const float* __restrict__ Ws1,
                           const float* __restrict__ bs1,
                           float* __restrict__ film) {
    const int j = blockIdx.x * blockDim.x + threadIdx.x;  // 8192
    const int b = j >> 10, col = j & 1023;
    const float* lat = latent + (size_t)b * DD;
    float s = bs1[col];
    for (int k = 0; k < DD; ++k) {
        const float x = lat[k];
        s += (x / (1.f + __expf(-x))) * Ws1[(size_t)k * 1024 + col];
    }
    film[j] = s;
}

// ---------------------------------------------------------------------------
// hs = bf16( silu( LN(ofc)*(1+scale)+shift ) ).  One wave per row (512 elems).
// ---------------------------------------------------------------------------
__global__ void film_apply_k(const float* __restrict__ ofc,
                             const float* __restrict__ film,
                             const float* __restrict__ en_g,
                             const float* __restrict__ en_b,
                             unsigned short* __restrict__ hs) {
    const int lane = threadIdx.x & 31;
    const int wave = threadIdx.x >> 5;
    const int r = blockIdx.x * 8 + wave;
    const int b = r >> 10;
    const float* p = ofc + (size_t)r * DD;

    float vals[16];
    float s = 0.f;
    #pragma unroll
    for (int i = 0; i < 16; ++i) { vals[i] = p[i * 32 + lane]; s += vals[i]; }
    const float mean = wave_sum(s) * (1.f / DD);
    float v2 = 0.f;
    #pragma unroll
    for (int i = 0; i < 16; ++i) { const float d = vals[i] - mean; v2 += d * d; }
    const float rstd = rsqrtf(wave_sum(v2) * (1.f / DD) + 1e-5f);

    #pragma unroll
    for (int i = 0; i < 16; ++i) {
        const int col = i * 32 + lane;
        const float ln = (vals[i] - mean) * rstd * en_g[col] + en_b[col];
        const float sc = film[(size_t)b * 1024 + col];
        const float sh = film[(size_t)b * 1024 + 512 + col];
        const float h  = ln * (1.f + sc) + sh;
        const float hz = h / (1.f + __expf(-h));
        hs[(size_t)r * DD + col] = f2bf(hz);
    }
}

// ---------------------------------------------------------------------------
// out = LN(y + residual), eps 1e-6. One wave per row.
// ---------------------------------------------------------------------------
__global__ void final_ln_k(const float* __restrict__ y,
                           const float* __restrict__ residual,
                           const float* __restrict__ ln_g,
                           const float* __restrict__ ln_b,
                           float* __restrict__ out) {
    const int lane = threadIdx.x & 31;
    const int wave = threadIdx.x >> 5;
    const int r = blockIdx.x * 8 + wave;
    const float* py = y + (size_t)r * DD;
    const float* pr = residual + (size_t)r * DD;

    float vals[16];
    float s = 0.f;
    #pragma unroll
    for (int i = 0; i < 16; ++i) {
        vals[i] = py[i * 32 + lane] + pr[i * 32 + lane];
        s += vals[i];
    }
    const float mean = wave_sum(s) * (1.f / DD);
    float v2 = 0.f;
    #pragma unroll
    for (int i = 0; i < 16; ++i) { const float d = vals[i] - mean; v2 += d * d; }
    const float rstd = rsqrtf(wave_sum(v2) * (1.f / DD) + 1e-6f);

    #pragma unroll
    for (int i = 0; i < 16; ++i) {
        const int col = i * 32 + lane;
        out[(size_t)r * DD + col] = (vals[i] - mean) * rstd * ln_g[col] + ln_b[col];
    }
}

// ---------------------------------------------------------------------------
extern "C" void kernel_launch(void* const* d_in, const int* in_sizes, int n_in,
                              void* d_out, int out_size, void* d_ws, size_t ws_size,
                              hipStream_t stream) {
    (void)in_sizes; (void)n_in; (void)out_size; (void)ws_size;

    const float* q      = (const float*)d_in[0];
    const float* k      = (const float*)d_in[1];
    const float* v      = (const float*)d_in[2];
    const float* latent = (const float*)d_in[3];
    const float* Wq  = (const float*)d_in[4];   const float* bq  = (const float*)d_in[5];
    const float* Wk  = (const float*)d_in[6];   const float* bk  = (const float*)d_in[7];
    const float* Wv  = (const float*)d_in[8];   const float* bv  = (const float*)d_in[9];
    const float* Wfc = (const float*)d_in[10];  const float* bfc = (const float*)d_in[11];
    const float* Ws1 = (const float*)d_in[12];  const float* bs1 = (const float*)d_in[13];
    const float* Ws2 = (const float*)d_in[14];  const float* bs2 = (const float*)d_in[15];
    const float* en_g = (const float*)d_in[16]; const float* en_b = (const float*)d_in[17];
    const float* ln_g = (const float*)d_in[18]; const float* ln_b = (const float*)d_in[19];

    float* out  = (float*)d_out;
    float* attn = out + (size_t)BB * LL * DD;              // [B,H,L,L]

    // workspace carve-up (all chunks 16B-aligned by construction)
    char* w = (char*)d_ws;
    const size_t SZ_BF = (size_t)BB * HH * LL * DKD * 2;   // 8 MB
    const size_t SZ_F  = (size_t)BB * LL * DD * 4;         // 16 MB
    const size_t SZ_W  = (size_t)512 * 512 * 2;            // 0.5 MB
    unsigned short* Qbf  = (unsigned short*)w;   w += SZ_BF;
    unsigned short* Kbf  = (unsigned short*)w;   w += SZ_BF;
    unsigned short* Vt   = (unsigned short*)w;   w += SZ_BF;
    unsigned short* xbf  = (unsigned short*)w;   w += SZ_BF;   // staging for X operands
    unsigned short* WqT  = (unsigned short*)w;   w += SZ_W;
    unsigned short* WkT  = (unsigned short*)w;   w += SZ_W;
    unsigned short* WvT  = (unsigned short*)w;   w += SZ_W;
    unsigned short* WfcT = (unsigned short*)w;   w += SZ_W;
    unsigned short* Ws2T = (unsigned short*)w;   w += SZ_W;
    float*          o    = (float*)w;            w += SZ_F;
    float*          ofc  = (float*)w;            w += SZ_F;
    unsigned short* hs   = (unsigned short*)w;   w += SZ_BF;
    float*          film = (float*)w;            w += (size_t)BB * 1024 * 4;
    float*          y    = o;   // o dead after Wfc GEMM; reuse for y

    const dim3 blk(256);
    const dim3 gGemm(8, 32);               // N/64 x M/256
    const dim3 gCvtX(2048);                // 4.19M elems / (256*8)
    const dim3 gCvtW(1024);                // 262144 / 256

    // 0: weight convert+transpose to bf16 [N,K]
    hipLaunchKernelGGL(cvt_w_t, gCvtW, blk, 0, stream, Wq,  WqT);
    hipLaunchKernelGGL(cvt_w_t, gCvtW, blk, 0, stream, Wk,  WkT);
    hipLaunchKernelGGL(cvt_w_t, gCvtW, blk, 0, stream, Wv,  WvT);
    hipLaunchKernelGGL(cvt_w_t, gCvtW, blk, 0, stream, Wfc, WfcT);
    hipLaunchKernelGGL(cvt_w_t, gCvtW, blk, 0, stream, Ws2, Ws2T);

    // 1-3: QKV projections; K/V stored in attention-friendly layouts
    hipLaunchKernelGGL(cvt_x_bf16, gCvtX, blk, 0, stream, q, xbf);
    hipLaunchKernelGGL((gemm512_wmma<1>), gGemm, blk, 0, stream, xbf, WqT, bq, (void*)Qbf);
    hipLaunchKernelGGL(cvt_x_bf16, gCvtX, blk, 0, stream, k, xbf);
    hipLaunchKernelGGL((gemm512_wmma<1>), gGemm, blk, 0, stream, xbf, WkT, bk, (void*)Kbf);
    hipLaunchKernelGGL(cvt_x_bf16, gCvtX, blk, 0, stream, v, xbf);
    hipLaunchKernelGGL((gemm512_wmma<2>), gGemm, blk, 0, stream, xbf, WvT, bv, (void*)Vt);

    // 4: scores -> attn region (pre-softmax)
    hipLaunchKernelGGL(attn_scores_wmma, dim3(BB * HH, LL / 32, 2), blk, 0, stream,
                       Qbf, Kbf, attn);
    // 5: softmax in place
    hipLaunchKernelGGL(softmax_rows_k, dim3((BB * HH * LL) / 8), blk, 0, stream, attn);
    // 6: o = attn @ V
    hipLaunchKernelGGL(attn_v_wmma, dim3(BB * HH, LL / 256), blk, 0, stream, attn, Vt, o);

    // 7: ofc = o @ Wfc + bfc
    hipLaunchKernelGGL(cvt_x_bf16, gCvtX, blk, 0, stream, o, xbf);
    hipLaunchKernelGGL((gemm512_wmma<0>), gGemm, blk, 0, stream, xbf, WfcT, bfc, (void*)ofc);

    // 8: FiLM scale/shift
    hipLaunchKernelGGL(film_mlp_k, dim3(32), blk, 0, stream, latent, Ws1, bs1, film);
    // 9: LN + FiLM + SiLU -> bf16
    hipLaunchKernelGGL(film_apply_k, dim3((BB * LL) / 8), blk, 0, stream, ofc, film, en_g, en_b, hs);
    // 10: y = hs @ Ws2 + bs2
    hipLaunchKernelGGL((gemm512_wmma<0>), gGemm, blk, 0, stream, hs, Ws2T, bs2, (void*)y);
    // 11: out = LN(y + q)
    hipLaunchKernelGGL(final_ln_k, dim3((BB * LL) / 8), blk, 0, stream, y, q, ln_g, ln_b, out);
}
